// SLatFlowModel_9431748182658
// MI455X (gfx1250) — matrix operand order repaired and checked by reading, not compiled, not run on hardware
//
#include <hip/hip_runtime.h>
#include <hip/hip_bf16.h>
#include <math.h>

// ---------------------------------------------------------------------------
// CDNA5 (gfx1250) WMMA helpers — wave32, V_WMMA_F32_16X16X32_F16
// ---------------------------------------------------------------------------
typedef __attribute__((ext_vector_type(16))) _Float16 v16h;
typedef __attribute__((ext_vector_type(8)))  _Float16 h8;
typedef __attribute__((ext_vector_type(4)))  _Float16 h4;
typedef __attribute__((ext_vector_type(8)))  float    v8f;
typedef __attribute__((ext_vector_type(4)))  float    f4;

#define NFp 8192
#define NCp 1024
#define MCp 1024
#define Hp  16
#define HDp 64
#define NBp 24

__device__ inline v8f zero8() {
  v8f z;
#pragma unroll
  for (int i = 0; i < 8; ++i) z[i] = 0.f;
  return z;
}

__device__ inline v8f wmma_f16(v16h a, v16h b, v8f c) {
  return __builtin_amdgcn_wmma_f32_16x16x32_f16(
      false, a, false, b, (short)0, c, false, false);
}

// A-matrix 16x32 f16 fragment from LDS, row-major [m][k], leading dim ld (halves).
// Per lane: k = hi*8+j (j<8) and 16+hi*8+(j-8) -> two contiguous 8-half (16B) runs.
__device__ inline v16h load_a_frag(const _Float16* s, int ld, int row0, int kbase) {
  int lane = threadIdx.x & 31;
  int m = lane & 15, hi = lane >> 4;
  const _Float16* p = s + (size_t)(row0 + m) * ld + kbase;
  union { v16h v; h8 h[2]; } u;
  u.h[0] = *(const h8*)(p + hi * 8);
  u.h[1] = *(const h8*)(p + 16 + hi * 8);
  return u.v;
}

// B-matrix 32x16 f16 fragment from LDS stored k-major: element (k,n) at s[n*ldk + k].
// Per lane: k = hi*8+j (j<8) and 16+hi*8+(j-8) -> two contiguous 8-half (16B) runs.
__device__ inline v16h load_b_frag_nk(const _Float16* s, int ldk, int kbase, int n0) {
  int lane = threadIdx.x & 31;
  int n = lane & 15, hi = lane >> 4;
  const _Float16* p = s + (size_t)(n0 + n) * ldk + kbase;
  union { v16h v; h8 h[2]; } u;
  u.h[0] = *(const h8*)(p + hi * 8);
  u.h[1] = *(const h8*)(p + 16 + hi * 8);
  return u.v;
}

// ---------------------------------------------------------------------------
// Generic GEMM: C[M,N] = A[M,K] @ B[K,N] (+bias) (+epi: 0=none 1=exact gelu)
// 128x64 tile, 8 waves; each wave: one 16-row stripe x 64 cols (4 WMMA tiles).
// ---------------------------------------------------------------------------
__global__ __launch_bounds__(256) void gemm_kernel(
    const float* __restrict__ A, const float* __restrict__ B,
    const float* __restrict__ bias, float* __restrict__ C,
    int M, int N, int K, int epi) {
  __shared__ __align__(16) _Float16 As[128 * 32];   // [m][k]
  __shared__ __align__(16) _Float16 Bs[64 * 32];    // [n][k]  (transposed)
  int tileM = blockIdx.x * 128, tileN = blockIdx.y * 64;
  int tid = threadIdx.x, wave = tid >> 5, lane = tid & 31;
  v8f acc[4] = {zero8(), zero8(), zero8(), zero8()};
  for (int k0 = 0; k0 < K; k0 += 32) {
    // stage A: 128x32 floats as f16, vectorized f4 -> h4
    for (int g = tid; g < 1024; g += 256) {
      int r = g >> 3, c4 = (g & 7) * 4;
      int gm = tileM + r, gk = k0 + c4;
      f4 v = {0.f, 0.f, 0.f, 0.f};
      if (gm < M && gk < K) v = *(const f4*)(A + (size_t)gm * K + gk);
      h4 h = {(_Float16)v[0], (_Float16)v[1], (_Float16)v[2], (_Float16)v[3]};
      *(h4*)(As + r * 32 + c4) = h;
    }
    // stage B transposed: Bs[n][k]; global reads coalesced along n
    for (int g = tid; g < 512; g += 256) {
      int n = g & 63, kq = g >> 6;
      int gn = tileN + n, kk = kq * 4;
      h4 h;
#pragma unroll
      for (int j = 0; j < 4; ++j) {
        int gk = k0 + kk + j;
        float v = (gk < K && gn < N) ? B[(size_t)gk * N + gn] : 0.f;
        h[j] = (_Float16)v;
      }
      *(h4*)(Bs + n * 32 + kk) = h;
    }
    if (k0 + 32 < K)  // prefetch next weight k-tile -> global_prefetch_b8
      __builtin_prefetch(B + (size_t)(k0 + 32) * N + tileN + (tid & 63), 0, 1);
    __syncthreads();
    v16h a = load_a_frag(As, 32, wave * 16, 0);
#pragma unroll
    for (int nt = 0; nt < 4; ++nt) {
      v16h b = load_b_frag_nk(Bs, 32, 0, nt * 16);
      acc[nt] = wmma_f16(a, b, acc[nt]);
    }
    __syncthreads();
  }
  int n16 = lane & 15, hi = lane >> 4;
#pragma unroll
  for (int nt = 0; nt < 4; ++nt) {
#pragma unroll
    for (int r = 0; r < 8; ++r) {
      int gm = tileM + wave * 16 + r + 8 * hi;
      int gn = tileN + nt * 16 + n16;
      if (gm < M && gn < N) {
        float v = acc[nt][r] + (bias ? bias[gn] : 0.f);
        if (epi == 1) v = 0.5f * v * (1.f + erff(v * 0.70710678f));
        C[(size_t)gm * N + gn] = v;
      }
    }
  }
}

// ---------------------------------------------------------------------------
// Sparse 27-tap conv: Out[M,cout] = bias + sum_tap gather(X, nbr[:,tap]) @ W[tap]
// Same 128x64 tiling; rows with index == M (sentinel) contribute zero.
// ---------------------------------------------------------------------------
__global__ __launch_bounds__(256) void spconv_kernel(
    const float* __restrict__ X, const int* __restrict__ nbr,
    const float* __restrict__ W, const float* __restrict__ bias,
    float* __restrict__ Out, int M, int cin, int cout) {
  __shared__ __align__(16) _Float16 As[128 * 32];
  __shared__ __align__(16) _Float16 Bs[64 * 32];
  __shared__ int idxs[128];
  int tileM = blockIdx.x * 128, tileN = blockIdx.y * 64;
  int tid = threadIdx.x, wave = tid >> 5, lane = tid & 31;
  v8f acc[4] = {zero8(), zero8(), zero8(), zero8()};
  for (int tap = 0; tap < 27; ++tap) {
    if (tid < 128) {
      int gm = tileM + tid;
      idxs[tid] = (gm < M) ? nbr[(size_t)gm * 27 + tap] : M;
    }
    __syncthreads();
    const float* Wt = W + (size_t)tap * cin * cout;
    for (int k0 = 0; k0 < cin; k0 += 32) {
      for (int g = tid; g < 1024; g += 256) {
        int r = g >> 3, c4 = (g & 7) * 4;
        int src = idxs[r], gk = k0 + c4;
        f4 v = {0.f, 0.f, 0.f, 0.f};
        if (src < M && gk < cin) v = *(const f4*)(X + (size_t)src * cin + gk);
        h4 h = {(_Float16)v[0], (_Float16)v[1], (_Float16)v[2], (_Float16)v[3]};
        *(h4*)(As + r * 32 + c4) = h;
      }
      for (int g = tid; g < 512; g += 256) {
        int n = g & 63, kq = g >> 6;
        int gn = tileN + n, kk = kq * 4;
        h4 h;
#pragma unroll
        for (int j = 0; j < 4; ++j) {
          int gk = k0 + kk + j;
          float v = (gk < cin && gn < cout) ? Wt[(size_t)gk * cout + gn] : 0.f;
          h[j] = (_Float16)v;
        }
        *(h4*)(Bs + n * 32 + kk) = h;
      }
      if (k0 + 32 < cin)
        __builtin_prefetch(Wt + (size_t)(k0 + 32) * cout + tileN + (tid & 63), 0, 1);
      __syncthreads();
      v16h a = load_a_frag(As, 32, wave * 16, 0);
#pragma unroll
      for (int nt = 0; nt < 4; ++nt) {
        v16h b = load_b_frag_nk(Bs, 32, 0, nt * 16);
        acc[nt] = wmma_f16(a, b, acc[nt]);
      }
      __syncthreads();
    }
  }
  int n16 = lane & 15, hi = lane >> 4;
#pragma unroll
  for (int nt = 0; nt < 4; ++nt) {
#pragma unroll
    for (int r = 0; r < 8; ++r) {
      int gm = tileM + wave * 16 + r + 8 * hi;
      int gn = tileN + nt * 16 + n16;
      if (gm < M && gn < cout)
        Out[(size_t)gm * cout + gn] = acc[nt][r] + (bias ? bias[gn] : 0.f);
    }
  }
}

// ---------------------------------------------------------------------------
// Flash attention (HD=64, wave32 WMMA): one block = 128 queries x 1 head.
// Each wave owns a 16-query stripe; key chunks of 32, online softmax.
// Ks staged [key][d] (contiguous d for K^T frags); Vs staged [d][key].
// ---------------------------------------------------------------------------
#define FAQ 128
__global__ __launch_bounds__(256) void flash_kernel(
    const float* __restrict__ Q, int qs,
    const float* __restrict__ K, int ks,
    const float* __restrict__ V, int vs,
    float* __restrict__ O, int os,
    int nq, int nk, float scale) {
  __shared__ __align__(16) _Float16 Qs[FAQ * 64];   // [q][d]
  __shared__ __align__(16) _Float16 Ks[32 * 64];    // [key][d]
  __shared__ __align__(16) _Float16 Vt[64 * 32];    // [d][key]
  __shared__ __align__(16) _Float16 Ps[8 * 16 * 32];
  int head = blockIdx.y;
  int q0 = blockIdx.x * FAQ;
  int tid = threadIdx.x, wave = tid >> 5, lane = tid & 31;
  int n16 = lane & 15, hi = lane >> 4;

  for (int g = tid; g < FAQ * 16; g += 256) {
    int r = g >> 4, c4 = (g & 15) * 4;
    int qm = q0 + r;
    f4 v = {0.f, 0.f, 0.f, 0.f};
    if (qm < nq) v = *(const f4*)(Q + (size_t)qm * qs + head * 64 + c4);
    h4 h = {(_Float16)v[0], (_Float16)v[1], (_Float16)v[2], (_Float16)v[3]};
    *(h4*)(Qs + r * 64 + c4) = h;
  }
  __syncthreads();
  v16h qa0 = load_a_frag(Qs, 64, wave * 16, 0);
  v16h qa1 = load_a_frag(Qs, 64, wave * 16, 32);

  v8f o0 = zero8(), o1 = zero8(), o2 = zero8(), o3 = zero8();
  float mrun[8], lrun[8];
#pragma unroll
  for (int r = 0; r < 8; ++r) { mrun[r] = -3.0e38f; lrun[r] = 0.f; }

  for (int kc = 0; kc < nk; kc += 32) {
    // K chunk: [key][d], vectorized
    for (int g = tid; g < 512; g += 256) {
      int r = g >> 4, c4 = (g & 15) * 4;
      f4 v = *(const f4*)(K + (size_t)(kc + r) * ks + head * 64 + c4);
      h4 h = {(_Float16)v[0], (_Float16)v[1], (_Float16)v[2], (_Float16)v[3]};
      *(h4*)(Ks + r * 64 + c4) = h;
    }
    // V chunk transposed: Vt[d][key]
    for (int g = tid; g < 512; g += 256) {
      int key = g >> 4, d4 = (g & 15) * 4;
      f4 v = *(const f4*)(V + (size_t)(kc + key) * vs + head * 64 + d4);
#pragma unroll
      for (int j = 0; j < 4; ++j) Vt[(d4 + j) * 32 + key] = (_Float16)v[j];
    }
    __syncthreads();
    // S[16 x 32] = Q(16x64) @ K^T(64x32): B element (k=d, n=key) at Ks[key*64+d]
    v8f s0v, s1v;
    {
      v16h b0 = load_b_frag_nk(Ks, 64, 0, 0);
      v16h b1 = load_b_frag_nk(Ks, 64, 32, 0);
      v8f z = zero8(); z = wmma_f16(qa0, b0, z); z = wmma_f16(qa1, b1, z); s0v = z;
    }
    {
      v16h b0 = load_b_frag_nk(Ks, 64, 0, 16);
      v16h b1 = load_b_frag_nk(Ks, 64, 32, 16);
      v8f z = zero8(); z = wmma_f16(qa0, b0, z); z = wmma_f16(qa1, b1, z); s1v = z;
    }
    float alpha[8];
#pragma unroll
    for (int r = 0; r < 8; ++r) {
      float v0 = s0v[r] * scale, v1 = s1v[r] * scale;
      float mx = fmaxf(v0, v1);
#pragma unroll
      for (int off = 1; off < 16; off <<= 1) mx = fmaxf(mx, __shfl_xor(mx, off, 16));
      float mnew = fmaxf(mrun[r], mx);
      float al = __expf(mrun[r] - mnew);
      mrun[r] = mnew; alpha[r] = al;
      float p0 = __expf(v0 - mnew), p1 = __expf(v1 - mnew);
      float rs = p0 + p1;
#pragma unroll
      for (int off = 1; off < 16; off <<= 1) rs += __shfl_xor(rs, off, 16);
      lrun[r] = lrun[r] * al + rs;
      Ps[wave * 512 + (r + 8 * hi) * 32 + n16] = (_Float16)p0;
      Ps[wave * 512 + (r + 8 * hi) * 32 + 16 + n16] = (_Float16)p1;
    }
#pragma unroll
    for (int r = 0; r < 8; ++r) {
      o0[r] *= alpha[r]; o1[r] *= alpha[r]; o2[r] *= alpha[r]; o3[r] *= alpha[r];
    }
    // O += P(16x32) @ V(32x64): B element (k=key, n=d) at Vt[d*32+key]
    v16h pa  = load_a_frag(Ps + wave * 512, 32, 0, 0);
    v16h vb0 = load_b_frag_nk(Vt, 32, 0, 0);
    v16h vb1 = load_b_frag_nk(Vt, 32, 0, 16);
    v16h vb2 = load_b_frag_nk(Vt, 32, 0, 32);
    v16h vb3 = load_b_frag_nk(Vt, 32, 0, 48);
    o0 = wmma_f16(pa, vb0, o0);
    o1 = wmma_f16(pa, vb1, o1);
    o2 = wmma_f16(pa, vb2, o2);
    o3 = wmma_f16(pa, vb3, o3);
    __syncthreads();
  }
#pragma unroll
  for (int r = 0; r < 8; ++r) {
    int qm = q0 + wave * 16 + r + 8 * hi;
    if (qm < nq) {
      float inv = 1.f / lrun[r];
      size_t base = (size_t)qm * os + head * 64;
      O[base + 0  + n16] = o0[r] * inv;
      O[base + 16 + n16] = o1[r] * inv;
      O[base + 32 + n16] = o2[r] * inv;
      O[base + 48 + n16] = o3[r] * inv;
    }
  }
}

// ---------------------------------------------------------------------------
// LayerNorm (+affine / +(1+scale),shift modulation / +SiLU). One block per row.
// ---------------------------------------------------------------------------
__global__ __launch_bounds__(256) void ln_kernel(
    const float* __restrict__ X, float* __restrict__ Y,
    const float* __restrict__ g, const float* __restrict__ b,
    const float* __restrict__ sc, const float* __restrict__ sh,
    int C, int act, float eps) {
  __shared__ float red[256];
  int row = blockIdx.x, tid = threadIdx.x;
  const float* x = X + (size_t)row * C;
  float* y = Y + (size_t)row * C;
  float s = 0.f;
  for (int c = tid; c < C; c += 256) s += x[c];
  red[tid] = s; __syncthreads();
  for (int o = 128; o > 0; o >>= 1) { if (tid < o) red[tid] += red[tid + o]; __syncthreads(); }
  float mean = red[0] / C; __syncthreads();
  float v = 0.f;
  for (int c = tid; c < C; c += 256) { float d = x[c] - mean; v += d * d; }
  red[tid] = v; __syncthreads();
  for (int o = 128; o > 0; o >>= 1) { if (tid < o) red[tid] += red[tid + o]; __syncthreads(); }
  float rstd = rsqrtf(red[0] / C + eps); __syncthreads();
  for (int c = tid; c < C; c += 256) {
    float t = (x[c] - mean) * rstd;
    if (g)  t = t * g[c] + b[c];
    if (sc) t = t * (1.f + sc[c]) + sh[c];
    if (act) t = t / (1.f + __expf(-t));
    y[c] = t;
  }
}

// Per-(token,head) RMS norm of q or k slice, in place. One wave per (row,head).
__global__ __launch_bounds__(256) void rmsqk_kernel(
    float* __restrict__ X, int stride, int ofs, const float* __restrict__ g, int rows) {
  int gid = blockIdx.x * 8 + (threadIdx.x >> 5);
  int lane = threadIdx.x & 31;
  int row = gid / Hp, head = gid % Hp;
  if (row >= rows) return;
  float* p = X + (size_t)row * stride + ofs + head * HDp;
  float v0 = p[lane], v1 = p[lane + 32];
  float s = v0 * v0 + v1 * v1;
#pragma unroll
  for (int o = 16; o > 0; o >>= 1) s += __shfl_xor(s, o, 32);
  float r = rsqrtf(s / HDp + 1e-6f);
  p[lane]      = v0 * r * g[head * HDp + lane];
  p[lane + 32] = v1 * r * g[head * HDp + lane + 32];
}

// GEMV: out[N] = act( a[K] @ W[K,N] + bias ).  act: 0=none 1=silu
__global__ void gemv_kernel(const float* __restrict__ a, const float* __restrict__ Wm,
                            const float* __restrict__ bias, float* __restrict__ out,
                            int K, int N, int act) {
  int n = blockIdx.x * 256 + threadIdx.x;
  if (n >= N) return;
  float acc = bias ? bias[n] : 0.f;
  for (int k = 0; k < K; ++k) acc = fmaf(a[k], Wm[(size_t)k * N + n], acc);
  if (act == 1) acc = acc / (1.f + __expf(-acc));
  out[n] = acc;
}

__global__ void silu_kernel(const float* __restrict__ x, float* __restrict__ y, int n) {
  int i = blockIdx.x * 256 + threadIdx.x;
  if (i < n) { float v = x[i]; y[i] = v / (1.f + __expf(-v)); }
}

// x[i] += (gate ? gate[i%C] : 1) * y[i]
__global__ void resadd_kernel(float* __restrict__ x, const float* __restrict__ y,
                              const float* __restrict__ gate, size_t n, int C) {
  size_t i = (size_t)blockIdx.x * 256 + threadIdx.x;
  if (i >= n) return;
  float g = gate ? gate[i % C] : 1.f;
  x[i] += g * y[i];
}

__global__ void add_kernel(const float* __restrict__ a, const float* __restrict__ b,
                           float* __restrict__ o, size_t n) {
  size_t i = (size_t)blockIdx.x * 256 + threadIdx.x;
  if (i < n) o[i] = a[i] + b[i];
}

// Out[r, 0:Ca]=A[src,·]  Out[r, Ca:Ca+Cb]=B[src,·]  src = parent ? parent[r] : r
__global__ void gathercat_kernel(const float* __restrict__ A, int Ca,
                                 const float* __restrict__ Bsrc, int Cb,
                                 const int* __restrict__ parent,
                                 float* __restrict__ Out, int rows) {
  int C = Ca + Cb;
  size_t i = (size_t)blockIdx.x * 256 + threadIdx.x;
  if (i >= (size_t)rows * C) return;
  int r = (int)(i / C), c = (int)(i % C);
  int sr = parent ? parent[r] : r;
  Out[i] = (c < Ca) ? A[(size_t)sr * Ca + c] : Bsrc[(size_t)sr * Cb + (c - Ca)];
}

__global__ void segadd_kernel(const float* __restrict__ X, const int* __restrict__ parent,
                              float* __restrict__ Sum, float* __restrict__ Cnt,
                              int rows, int C) {
  size_t i = (size_t)blockIdx.x * 256 + threadIdx.x;
  if (i >= (size_t)rows * C) return;
  int r = (int)(i / C), c = (int)(i % C);
  atomicAdd(&Sum[(size_t)parent[r] * C + c], X[i]);
  if (c == 0) atomicAdd(&Cnt[parent[r]], 1.f);
}

__global__ void segdiv_kernel(float* __restrict__ Sum, const float* __restrict__ Cnt,
                              int nout, int C) {
  size_t i = (size_t)blockIdx.x * 256 + threadIdx.x;
  if (i >= (size_t)nout * C) return;
  Sum[i] /= fmaxf(Cnt[i / C], 1.f);
}

// hc = s1 + sinusoidal positional embedding of integer coords (fd = 170, pad to 1024)
__global__ void posadd_kernel(const float* __restrict__ S1, const int* __restrict__ coords,
                              float* __restrict__ Out, int rows) {
  size_t i = (size_t)blockIdx.x * 256 + threadIdx.x;
  if (i >= (size_t)rows * MCp) return;
  int r = (int)(i >> 10), c = (int)(i & 1023);
  float pe = 0.f;
  if (c < 1020) {
    int j = c / 340, rr = c % 340;
    int f = rr % 170;
    float freq = __expf(-logf(10000.f) * (float)f / 170.f);
    float a = (float)coords[r * 3 + j] * freq;
    pe = (rr < 170) ? sinf(a) : cosf(a);
  }
  Out[i] = S1[i] + pe;
}

// e[0:128]=cos(t*f_j), e[128:256]=sin(t*f_j), f_j = exp(-ln(1e4)*j/128)
__global__ void tembed_base_kernel(const float* __restrict__ t, float* __restrict__ e) {
  int j = threadIdx.x;
  if (j < 128) {
    float freq = __expf(-logf(10000.f) * (float)j / 128.f);
    float a = t[0] * freq;
    e[j] = cosf(a);
    e[128 + j] = sinf(a);
  }
}

// ---------------------------------------------------------------------------
// Host orchestration
// ---------------------------------------------------------------------------
extern "C" void kernel_launch(void* const* d_in, const int* in_sizes, int n_in,
                              void* d_out, int out_size, void* d_ws, size_t ws_size,
                              hipStream_t stream) {
  (void)in_sizes; (void)n_in; (void)out_size; (void)ws_size;

  const float* x_feats = (const float*)d_in[0];
  const float* t_in    = (const float*)d_in[1];
  const float* cond    = (const float*)d_in[2];
  const int*   coords  = (const int*)d_in[3];
  const int*   nbrF    = (const int*)d_in[4];
  const int*   nbrC    = (const int*)d_in[5];
  const int*   parent  = (const int*)d_in[6];
  int pi = 7;
  auto nxt = [&]() { return (const float*)d_in[pi++]; };
  const float* in_W = nxt(); const float* in_b = nxt();
  const float* t_W1 = nxt(); const float* t_b1 = nxt();
  const float* t_W2 = nxt(); const float* t_b2 = nxt();
  struct RBp { const float *n1_g,*n1_b,*c1_W,*c1_b,*emb_W,*emb_b,*c2_W,*c2_b,*sk_W,*sk_b; };
  auto loadRB = [&](bool sk) {
    RBp r;
    r.n1_g = nxt(); r.n1_b = nxt(); r.c1_W = nxt(); r.c1_b = nxt();
    r.emb_W = nxt(); r.emb_b = nxt(); r.c2_W = nxt(); r.c2_b = nxt();
    if (sk) { r.sk_W = nxt(); r.sk_b = nxt(); } else { r.sk_W = nullptr; r.sk_b = nullptr; }
    return r;
  };
  RBp ib1 = loadRB(false);
  RBp ib2 = loadRB(true);
  const float* ada_W = nxt(); const float* ada_b = nxt();
  const float* qkv_W = nxt(); const float* qkv_b = nxt();
  const float* o_W   = nxt(); const float* o_b   = nxt();
  const float* qg    = nxt(); const float* kg    = nxt();
  const float* n2_g  = nxt(); const float* n2_b  = nxt();
  const float* cq_W  = nxt(); const float* cq_b  = nxt();
  const float* ckv_W = nxt(); const float* ckv_b = nxt();
  const float* co_W  = nxt(); const float* co_b  = nxt();
  const float* m1_W  = nxt(); const float* m1_b  = nxt();
  const float* m2_W  = nxt(); const float* m2_b  = nxt();
  RBp ob1 = loadRB(true);
  RBp ob2 = loadRB(true);
  const float* out_W = nxt(); const float* out_b = nxt();

  float* W = (float*)d_ws;
  size_t off = 0;
  auto alloc = [&](size_t n) { float* p = W + off; off += n; return p; };
  float* e256 = alloc(256);
  float* te   = alloc(MCp);
  float* te_s = alloc(MCp);
  float* embo = alloc(2048);
  float* mod  = alloc(6 * MCp);
  float* cnt  = alloc(NCp);
  float* f1   = alloc((size_t)NFp * 128);
  float* f2   = alloc((size_t)NFp * 128);
  float* f3   = alloc((size_t)NFp * 128);
  float* s0   = alloc((size_t)NFp * 128);
  float* xd   = alloc((size_t)NCp * 128);
  float* s1   = alloc((size_t)NCp * MCp);
  float* hc   = alloc((size_t)NCp * MCp);
  float* t2   = alloc((size_t)NCp * MCp);
  float* qkvb = alloc((size_t)NCp * 3 * MCp);
  float* t1   = alloc((size_t)NCp * 4 * MCp);
  float* att  = alloc((size_t)NCp * MCp);
  float* xup  = alloc((size_t)NFp * 2 * MCp);

  auto gemm = [&](const float* A, const float* B, const float* bias, float* C,
                  int M, int N, int K, int epi) {
    dim3 g((M + 127) / 128, (N + 63) / 64);
    gemm_kernel<<<g, 256, 0, stream>>>(A, B, bias, C, M, N, K, epi);
  };
  auto spconv = [&](const float* X, const int* nbr, const float* Wt, const float* bias,
                    float* Outp, int M, int cin, int cout) {
    dim3 g((M + 127) / 128, (cout + 63) / 64);
    spconv_kernel<<<g, 256, 0, stream>>>(X, nbr, Wt, bias, Outp, M, cin, cout);
  };
  auto gemv = [&](const float* a, const float* Wm, const float* bias, float* outp,
                  int K, int N, int act) {
    gemv_kernel<<<(N + 255) / 256, 256, 0, stream>>>(a, Wm, bias, outp, K, N, act);
  };
  auto ln = [&](const float* X, float* Y, const float* g, const float* b,
                const float* sc, const float* sh, int rows, int C, int act, float eps) {
    ln_kernel<<<rows, 256, 0, stream>>>(X, Y, g, b, sc, sh, C, act, eps);
  };
  auto addip = [&](float* x, const float* y, const float* gate, size_t rows, int C) {
    size_t n = rows * (size_t)C;
    resadd_kernel<<<(unsigned)((n + 255) / 256), 256, 0, stream>>>(x, y, gate, n, C);
  };
  auto flash = [&](const float* Q, int qs, const float* K, int ks, const float* V, int vs,
                   float* O, int os, int nq, int nk) {
    dim3 g(nq / FAQ, Hp);
    flash_kernel<<<g, 256, 0, stream>>>(Q, qs, K, ks, V, vs, O, os, nq, nk, 0.125f);
  };

  // ================= pipeline =================
  gemm(x_feats, in_W, in_b, f1, NFp, 128, 8, 0);

  tembed_base_kernel<<<1, 128, 0, stream>>>(t_in, e256);
  gemv(e256, t_W1, t_b1, embo, 256, MCp, 1);
  gemv(embo, t_W2, t_b2, te, MCp, MCp, 0);
  silu_kernel<<<(MCp + 255) / 256, 256, 0, stream>>>(te, te_s, MCp);

  // ---- ib1: resblock(NF, 128 -> 128), nbr_fine
  gemv(te_s, ib1.emb_W, ib1.emb_b, embo, MCp, 256, 0);
  ln(f1, f2, ib1.n1_g, ib1.n1_b, nullptr, nullptr, NFp, 128, 1, 1e-6f);
  spconv(f2, nbrF, ib1.c1_W, ib1.c1_b, f3, NFp, 128, 128);
  ln(f3, f3, nullptr, nullptr, embo, embo + 128, NFp, 128, 1, 1e-6f);
  spconv(f3, nbrF, ib1.c2_W, ib1.c2_b, f2, NFp, 128, 128);
  add_kernel<<<(unsigned)(((size_t)NFp * 128 + 255) / 256), 256, 0, stream>>>(
      f2, f1, s0, (size_t)NFp * 128);

  // ---- ib2: resblock(NC, 128 -> 1024), mode 'down', nbr_coarse
  gemv(te_s, ib2.emb_W, ib2.emb_b, embo, MCp, 2048, 0);
  hipMemsetAsync(xd, 0, (size_t)NCp * 128 * sizeof(float), stream);
  hipMemsetAsync(cnt, 0, NCp * sizeof(float), stream);
  segadd_kernel<<<(unsigned)(((size_t)NFp * 128 + 255) / 256), 256, 0, stream>>>(
      s0, parent, xd, cnt, NFp, 128);
  segdiv_kernel<<<(unsigned)(((size_t)NCp * 128 + 255) / 256), 256, 0, stream>>>(
      xd, cnt, NCp, 128);
  gemm(xd, ib2.sk_W, ib2.sk_b, s1, NCp, MCp, 128, 0);
  ln(xd, xd, ib2.n1_g, ib2.n1_b, nullptr, nullptr, NCp, 128, 1, 1e-6f);
  spconv(xd, nbrC, ib2.c1_W, ib2.c1_b, t2, NCp, 128, MCp);
  ln(t2, t2, nullptr, nullptr, embo, embo + MCp, NCp, MCp, 1, 1e-6f);
  spconv(t2, nbrC, ib2.c2_W, ib2.c2_b, att, NCp, MCp, MCp);
  addip(s1, att, nullptr, NCp, MCp);

  posadd_kernel<<<(unsigned)(((size_t)NCp * MCp + 255) / 256), 256, 0, stream>>>(
      s1, coords, hc, NCp);

  // ---- 24 transformer blocks
  for (int bi = 0; bi < NBp; ++bi) {
    const float* adaWb = ada_W + (size_t)bi * MCp * 6 * MCp;
    const float* adabb = ada_b + (size_t)bi * 6 * MCp;
    const float* qkvWb = qkv_W + (size_t)bi * MCp * 3 * MCp;
    const float* qkvbb = qkv_b + (size_t)bi * 3 * MCp;
    const float* oWb   = o_W  + (size_t)bi * MCp * MCp;
    const float* obb   = o_b  + (size_t)bi * MCp;
    const float* qgb   = qg   + (size_t)bi * Hp * HDp;
    const float* kgb   = kg   + (size_t)bi * Hp * HDp;
    const float* n2gb  = n2_g + (size_t)bi * MCp;
    const float* n2bb  = n2_b + (size_t)bi * MCp;
    const float* cqWb  = cq_W + (size_t)bi * MCp * MCp;
    const float* cqbb  = cq_b + (size_t)bi * MCp;
    const float* ckvWb = ckv_W + (size_t)bi * MCp * 2 * MCp;
    const float* ckvbb = ckv_b + (size_t)bi * 2 * MCp;
    const float* coWb  = co_W + (size_t)bi * MCp * MCp;
    const float* cobb  = co_b + (size_t)bi * MCp;
    const float* m1Wb  = m1_W + (size_t)bi * MCp * 4 * MCp;
    const float* m1bb  = m1_b + (size_t)bi * 4 * MCp;
    const float* m2Wb  = m2_W + (size_t)bi * 4 * MCp * MCp;
    const float* m2bb  = m2_b + (size_t)bi * MCp;

    gemv(te_s, adaWb, adabb, mod, MCp, 6 * MCp, 0);
    const float* sm = mod, *scm = mod + MCp, *gm = mod + 2 * MCp;
    const float* sf = mod + 3 * MCp, *scf = mod + 4 * MCp, *gf = mod + 5 * MCp;

    // self attention
    ln(hc, t2, nullptr, nullptr, scm, sm, NCp, MCp, 0, 1e-6f);
    gemm(t2, qkvWb, qkvbb, qkvb, NCp, 3 * MCp, MCp, 0);
    rmsqk_kernel<<<NCp * Hp / 8, 256, 0, stream>>>(qkvb, 3 * MCp, 0,   qgb, NCp);
    rmsqk_kernel<<<NCp * Hp / 8, 256, 0, stream>>>(qkvb, 3 * MCp, MCp, kgb, NCp);
    flash(qkvb, 3 * MCp, qkvb + MCp, 3 * MCp, qkvb + 2 * MCp, 3 * MCp,
          att, MCp, NCp, NCp);
    gemm(att, oWb, obb, t2, NCp, MCp, MCp, 0);
    addip(hc, t2, gm, NCp, MCp);

    // cross attention
    ln(hc, t2, n2gb, n2bb, nullptr, nullptr, NCp, MCp, 0, 1e-6f);
    gemm(t2, cqWb, cqbb, t1, NCp, MCp, MCp, 0);
    gemm(cond, ckvWb, ckvbb, qkvb, NCp, 2 * MCp, MCp, 0);
    flash(t1, MCp, qkvb, 2 * MCp, qkvb + MCp, 2 * MCp, att, MCp, NCp, NCp);
    gemm(att, coWb, cobb, t2, NCp, MCp, MCp, 0);
    addip(hc, t2, nullptr, NCp, MCp);

    // MLP
    ln(hc, t2, nullptr, nullptr, scf, sf, NCp, MCp, 0, 1e-6f);
    gemm(t2, m1Wb, m1bb, t1, NCp, 4 * MCp, MCp, 1);
    gemm(t1, m2Wb, m2bb, t2, NCp, MCp, 4 * MCp, 0);
    addip(hc, t2, gf, NCp, MCp);
  }

  // ---- ob1: x = up(concat([hc, s1])) -> resblock(NF, 2048 -> 128), nbr_fine
  gathercat_kernel<<<(unsigned)(((size_t)NFp * 2048 + 255) / 256), 256, 0, stream>>>(
      hc, MCp, s1, MCp, parent, xup, NFp);
  gemv(te_s, ob1.emb_W, ob1.emb_b, embo, MCp, 256, 0);
  gemm(xup, ob1.sk_W, ob1.sk_b, f1, NFp, 128, 2048, 0);
  ln(xup, xup, ob1.n1_g, ob1.n1_b, nullptr, nullptr, NFp, 2048, 1, 1e-6f);
  spconv(xup, nbrF, ob1.c1_W, ob1.c1_b, f2, NFp, 2048, 128);
  ln(f2, f2, nullptr, nullptr, embo, embo + 128, NFp, 128, 1, 1e-6f);
  spconv(f2, nbrF, ob1.c2_W, ob1.c2_b, f3, NFp, 128, 128);
  addip(f1, f3, nullptr, NFp, 128);

  // ---- ob2: x = concat([f1, s0]) -> resblock(NF, 256 -> 128), nbr_fine
  gathercat_kernel<<<(unsigned)(((size_t)NFp * 256 + 255) / 256), 256, 0, stream>>>(
      f1, 128, s0, 128, nullptr, xup, NFp);
  gemv(te_s, ob2.emb_W, ob2.emb_b, embo, MCp, 256, 0);
  gemm(xup, ob2.sk_W, ob2.sk_b, f2, NFp, 128, 256, 0);
  ln(xup, xup, ob2.n1_g, ob2.n1_b, nullptr, nullptr, NFp, 256, 1, 1e-6f);
  spconv(xup, nbrF, ob2.c1_W, ob2.c1_b, f3, NFp, 256, 128);
  ln(f3, f3, nullptr, nullptr, embo, embo + 128, NFp, 128, 1, 1e-6f);
  spconv(f3, nbrF, ob2.c2_W, ob2.c2_b, f1, NFp, 128, 128);
  addip(f1, f2, nullptr, NFp, 128);

  // ---- final LN (eps 1e-5, no affine) + out-proj
  ln(f1, f2, nullptr, nullptr, nullptr, nullptr, NFp, 128, 0, 1e-5f);
  gemm(f2, out_W, out_b, (float*)d_out, NFp, 8, 128, 0);
}
